// leo_gnn_28295244546760
// MI455X (gfx1250) — compile-verified, matrix operand order checked by reading, not run on hardware
//
#include <hip/hip_runtime.h>
#include <hip/hip_bf16.h>
#include <math.h>

typedef __attribute__((ext_vector_type(16))) _Float16 v16h;
typedef __attribute__((ext_vector_type(8)))  float    v8f;
typedef __attribute__((ext_vector_type(4)))  unsigned int u32x4;
typedef __attribute__((ext_vector_type(4)))  _Float16 h4;

union Frag {
    v16h v;
    u32x4 q[2];
    _Float16 h[16];
};

#define N_NODES 100000
#define N_ADJ   3200000
#define N_SCORE 1000000
#define FDIM    256

// ---------------------------------------------------------------------------
// Fragment loaders for V_WMMA_F32_16X16X32_F16 (wave32).
// A (16x32 f16, row-major source): lane L holds row M=L%16; half h=L>>4 selects
// K groups {h*8 .. h*8+7} and {16+h*8 .. 16+h*8+7}  -> two contiguous 16B loads.
// B (32x16 f16): pre-packed so each lane's 16 halves are contiguous (32B).
// ---------------------------------------------------------------------------
__device__ __forceinline__ v16h load_a_frag(const _Float16* __restrict__ rowptr,
                                            int kbase, int h) {
    Frag f;
    f.q[0] = *(const u32x4*)(rowptr + kbase + h * 8);
    f.q[1] = *(const u32x4*)(rowptr + kbase + 16 + h * 8);
    return f.v;
}

__device__ __forceinline__ v16h load_b_frag(const _Float16* __restrict__ Bp,
                                            int kt, int nt, int lane) {
    const _Float16* p = Bp + (((size_t)((kt * 16 + nt) * 32 + lane)) << 4);
    Frag f;
    f.q[0] = *(const u32x4*)(p);
    f.q[1] = *(const u32x4*)(p + 8);
    return f.v;
}

// ---------------------------------------------------------------------------
// Pack 256x256 f32 weights into per-lane-contiguous WMMA-B fragment layout,
// converting to f16. Also builds sym_w = 0.5*(W + W^T) for the scoring GEMM.
// Inverse of the B layout: element (k,n): kt=k/32, kk=k%32; g=kk>>4, r=kk&15,
// h=r>>3, q=r&7, w=q>>1, p=q&1; vgpr i=g*4+w; half slot e=2i+p; lane=h*16+n%16.
// ---------------------------------------------------------------------------
__global__ void k_pack_weights(const float* __restrict__ W1,
                               const float* __restrict__ W2,
                               const float* __restrict__ WL,
                               _Float16* __restrict__ W1p,
                               _Float16* __restrict__ W2p,
                               _Float16* __restrict__ Swp) {
    int idx = blockIdx.x * blockDim.x + threadIdx.x;
    if (idx >= FDIM * FDIM) return;
    int k = idx >> 8, n = idx & 255;
    int kt = k >> 5, kk = k & 31;
    int nt = n >> 4, nn = n & 15;
    int g = kk >> 4, r = kk & 15;
    int h = r >> 3, q = r & 7, w = q >> 1, p = q & 1;
    int i = g * 4 + w;
    int e = 2 * i + p;
    int lane = h * 16 + nn;
    size_t off = (((size_t)((kt * 16 + nt) * 32 + lane)) << 4) + e;
    W1p[off] = (_Float16)W1[idx];
    W2p[off] = (_Float16)W2[idx];
    Swp[off] = (_Float16)(0.5f * (WL[idx] + WL[n * FDIM + k]));
}

// f32 -> f16 convert, 4 elements/thread
__global__ void k_f32_to_f16(const float* __restrict__ in,
                             _Float16* __restrict__ out, long long n) {
    long long j = (((long long)blockIdx.x * blockDim.x) + threadIdx.x) * 4;
    if (j >= n) return;
    float4 v = *(const float4*)(in + j);
    h4 o;
    o.x = (_Float16)v.x; o.y = (_Float16)v.y;
    o.z = (_Float16)v.z; o.w = (_Float16)v.w;
    *(h4*)(out + j) = o;
}

// sum of 256-wide bias vector -> single scalar
__global__ void k_bias_sum(const float* __restrict__ b, float* __restrict__ out) {
    __shared__ float s[256];
    int t = threadIdx.x;
    s[t] = b[t];
    __syncthreads();
    for (int st = 128; st > 0; st >>= 1) {
        if (t < st) s[t] += s[t + st];
        __syncthreads();
    }
    if (t == 0) *out = s[0];
}

// ---------------------------------------------------------------------------
// C[M x 256] = A[M x 256, f16 row-major] @ B[256 x 256, pre-packed f16]
// 512 threads = 16 waves; wave nt owns output N-tile nt; block owns M-tile.
// ---------------------------------------------------------------------------
__global__ void __launch_bounds__(512) k_gemm_f16(const _Float16* __restrict__ A,
                                                  const _Float16* __restrict__ Bp,
                                                  float* __restrict__ C, int M) {
    int tid  = threadIdx.x;
    int lane = tid & 31;
    int nt   = tid >> 5;           // 0..15
    int mtile = blockIdx.x;
    if (mtile * 16 >= M) return;
    int h  = lane >> 4;
    int mr = lane & 15;
    const _Float16* arow = A + (size_t)(mtile * 16 + mr) * FDIM;

    v8f acc = {};
#pragma unroll
    for (int kt = 0; kt < 8; ++kt) {
        v16h a = load_a_frag(arow, kt * 32, h);
        v16h b = load_b_frag(Bp, kt, nt, lane);
        acc = __builtin_amdgcn_wmma_f32_16x16x32_f16(
            false, a, false, b, (short)0, acc, false, false);
    }
    int n = nt * 16 + mr;
#pragma unroll
    for (int i = 0; i < 8; ++i) {
        int m = i + 8 * h;  // D layout: vgpr i, lane-half h -> row m
        C[(size_t)(mtile * 16 + m) * FDIM + n] = acc[i];
    }
}

// ---------------------------------------------------------------------------
// agg[recv] += val * support[sender]  (64 threads per edge, float4 each)
// L2-resident tables -> native f32 atomics at L2.
// ---------------------------------------------------------------------------
__global__ void k_edge_agg(const float* __restrict__ support,
                           const int* __restrict__ senders,
                           const int* __restrict__ receivers,
                           const float* __restrict__ vals,
                           float* __restrict__ agg, int n_adj) {
    long long gid = (long long)blockIdx.x * blockDim.x + threadIdx.x;
    if (gid >= (long long)n_adj * 64) return;
    int e = (int)(gid >> 6);
    int c = ((int)gid & 63) << 2;
    int s = senders[e];
    int rcv = receivers[e];
    float v = vals[e];
    float4 sv = *(const float4*)(support + (size_t)s * FDIM + c);
    float* dst = agg + (size_t)rcv * FDIM + c;
    unsafeAtomicAdd(dst + 0, v * sv.x);
    unsafeAtomicAdd(dst + 1, v * sv.y);
    unsafeAtomicAdd(dst + 2, v * sv.z);
    unsafeAtomicAdd(dst + 3, v * sv.w);
}

// out_f16 = tanh(agg + bias[col]) , 4 elements/thread
__global__ void k_tanh_bias_h(const float* __restrict__ agg,
                              const float* __restrict__ bias,
                              _Float16* __restrict__ out, long long n) {
    long long j = (((long long)blockIdx.x * blockDim.x) + threadIdx.x) * 4;
    if (j >= n) return;
    float4 v = *(const float4*)(agg + j);
    int col = (int)(j & (FDIM - 1));
    h4 o;
    o.x = (_Float16)tanhf(v.x + bias[col + 0]);
    o.y = (_Float16)tanhf(v.y + bias[col + 1]);
    o.z = (_Float16)tanhf(v.z + bias[col + 2]);
    o.w = (_Float16)tanhf(v.w + bias[col + 3]);
    *(h4*)(out + j) = o;
}

// ---------------------------------------------------------------------------
// Fused scoring: per wave, 16 edges. t = e0 @ sym_w via WMMA (A fragments
// hoisted across the 16 N-tiles), then elementwise * e1 using the known D
// layout, cross-lane reduce (wave32 shfl_xor), + sum(bias), sigmoid.
// ---------------------------------------------------------------------------
__global__ void __launch_bounds__(256) k_score(const _Float16* __restrict__ embh,
                                               const _Float16* __restrict__ Swp,
                                               const int* __restrict__ edges,
                                               const float* __restrict__ biassum,
                                               float* __restrict__ out, int n_edges) {
    int lane = threadIdx.x & 31;
    long long wave = (long long)blockIdx.x * (blockDim.x >> 5) + (threadIdx.x >> 5);
    long long base = wave * 16;
    if (base >= n_edges) return;
    int h  = lane >> 4;
    int mr = lane & 15;
    long long myedge = base + mr;
    int r0 = edges[2 * myedge];
    int r1 = edges[2 * myedge + 1];

    const _Float16* arow = embh + (size_t)r0 * FDIM;
    v16h afr[8];
#pragma unroll
    for (int kt = 0; kt < 8; ++kt) afr[kt] = load_a_frag(arow, kt * 32, h);

    float sum[8];
#pragma unroll
    for (int i = 0; i < 8; ++i) sum[i] = 0.0f;

    for (int nt = 0; nt < 16; ++nt) {
        v8f acc = {};
#pragma unroll
        for (int kt = 0; kt < 8; ++kt) {
            v16h b = load_b_frag(Swp, kt, nt, lane);
            acc = __builtin_amdgcn_wmma_f32_16x16x32_f16(
                false, afr[kt], false, b, (short)0, acc, false, false);
        }
        int n = nt * 16 + mr;
#pragma unroll
        for (int i = 0; i < 8; ++i) {
            int m = i + 8 * h;                       // which edge this D row is
            int r1m = __shfl(r1, m, 32);             // broadcast its e1 row id
            float e1v = (float)embh[(size_t)r1m * FDIM + n];
            sum[i] += acc[i] * e1v;
        }
    }
    // reduce over the 16 N-lanes inside each half (masks 1,2,4,8 stay in-half)
#pragma unroll
    for (int i = 0; i < 8; ++i) {
#pragma unroll
        for (int mask = 1; mask < 16; mask <<= 1)
            sum[i] += __shfl_xor(sum[i], mask, 32);
    }
    float bs = *biassum;
    if (mr == 0) {  // lane 0 -> edges m=0..7 ; lane 16 -> edges m=8..15
#pragma unroll
        for (int i = 0; i < 8; ++i) {
            float s = sum[i] + bs;
            out[base + i + 8 * h] = 1.0f / (1.0f + expf(-s));
        }
    }
}

// ---------------------------------------------------------------------------
extern "C" void kernel_launch(void* const* d_in, const int* in_sizes, int n_in,
                              void* d_out, int out_size, void* d_ws, size_t ws_size,
                              hipStream_t stream) {
    const float* norm_features = (const float*)d_in[0];
    const int*   adj_senders   = (const int*)d_in[1];
    const int*   adj_receivers = (const int*)d_in[2];
    const float* adj_vals      = (const float*)d_in[3];
    const int*   edges         = (const int*)d_in[4];
    const float* W1            = (const float*)d_in[5];
    const float* b1            = (const float*)d_in[6];
    const float* W2            = (const float*)d_in[7];
    const float* b2            = (const float*)d_in[8];
    const float* weight_lin    = (const float*)d_in[9];
    const float* bias_lin      = (const float*)d_in[10];
    float* out = (float*)d_out;

    const size_t nn = N_NODES;
    const long long nelem = (long long)nn * FDIM;   // 25.6M

    // workspace carve-up (all offsets 16B-aligned)
    char* ws = (char*)d_ws;
    _Float16* xh   = (_Float16*)(ws);                              // f16 activations A
    _Float16* h1   = (_Float16*)(ws + nelem * 2);                  // f16 activations B
    float*    sup  = (float*)   (ws + nelem * 4);                  // f32 support
    float*    agg  = (float*)   (ws + nelem * 4 + nelem * 4);      // f32 aggregate
    char* wbase = ws + nelem * 12;
    _Float16* W1p  = (_Float16*)(wbase);
    _Float16* W2p  = (_Float16*)(wbase + FDIM * FDIM * 2);
    _Float16* Swp  = (_Float16*)(wbase + FDIM * FDIM * 4);
    float*    bsum = (float*)   (wbase + FDIM * FDIM * 6);

    // 1) pack weights (+ sym_w), convert X to f16, sum bias_lin
    k_pack_weights<<<(FDIM * FDIM + 255) / 256, 256, 0, stream>>>(
        W1, W2, weight_lin, W1p, W2p, Swp);
    k_f32_to_f16<<<(unsigned)((nelem / 4 + 255) / 256), 256, 0, stream>>>(
        norm_features, xh, nelem);
    k_bias_sum<<<1, 256, 0, stream>>>(bias_lin, bsum);

    const unsigned gemm_blocks = (unsigned)(nn / 16);                 // 6250
    const unsigned agg_blocks  = (unsigned)(((long long)N_ADJ * 64 + 255) / 256);
    const unsigned ew_blocks   = (unsigned)((nelem / 4 + 255) / 256);

    // 2) layer 1: support = Xh @ W1 ; agg = scatter-add ; x = tanh(agg+b1)
    k_gemm_f16<<<gemm_blocks, 512, 0, stream>>>(xh, W1p, sup, (int)nn);
    hipMemsetAsync(agg, 0, (size_t)nelem * 4, stream);
    k_edge_agg<<<agg_blocks, 256, 0, stream>>>(sup, adj_senders, adj_receivers,
                                               adj_vals, agg, N_ADJ);
    k_tanh_bias_h<<<ew_blocks, 256, 0, stream>>>(agg, b1, h1, nelem);

    // 3) layer 2: support = x @ W2 ; agg ; emb = tanh(agg+b2) (reuse xh buffer)
    k_gemm_f16<<<gemm_blocks, 512, 0, stream>>>(h1, W2p, sup, (int)nn);
    hipMemsetAsync(agg, 0, (size_t)nelem * 4, stream);
    k_edge_agg<<<agg_blocks, 256, 0, stream>>>(sup, adj_senders, adj_receivers,
                                               adj_vals, agg, N_ADJ);
    k_tanh_bias_h<<<ew_blocks, 256, 0, stream>>>(agg, b2, xh, nelem);

    // 4) fused scoring: 16 edges/wave, 8 waves/block
    const unsigned score_blocks = (unsigned)((N_SCORE + 127) / 128);
    k_score<<<score_blocks, 256, 0, stream>>>(xh, Swp, edges, bsum, out, N_SCORE);
}
